// CrossAttention1d_9569187135755
// MI455X (gfx1250) — compile-verified, matrix operand order checked by reading, not run on hardware
//
#include <hip/hip_runtime.h>
#include <hip/hip_bf16.h>
#include <math.h>

// ---------------------------------------------------------------------------
// CrossAttention1d for MI455X (gfx1250): bf16 WMMA pipeline, f32 accumulate.
// N=4, C=512, S=2048, S2=2048, H=8, CTX=512.  ~412 GFLOP -> compute-bound,
// all 5 matmul stages on v_wmma_f32_16x16x32_bf16, tiles staged to LDS via
// CDNA5 async copies (global_load_async_to_lds_b128 + s_wait_asynccnt).
// ---------------------------------------------------------------------------

typedef __bf16 bf16;
typedef __attribute__((ext_vector_type(16))) __bf16 v16bf;
typedef __attribute__((ext_vector_type(8)))  float  v8f;
typedef __attribute__((ext_vector_type(4)))  float  f4;

static constexpr int kN = 4, kC = 512, kS = 2048, kS2 = 2048, kH = 8, kCTX = 512;
static constexpr int kO = kC * kH;      // 4096
static constexpr float kEPS = 1e-5f;

__device__ __forceinline__ bf16 f2bf(float f) {
  unsigned u = __builtin_bit_cast(unsigned, f);
  u += 0x7FFFu + ((u >> 16) & 1u);          // round-to-nearest-even
  unsigned short h = (unsigned short)(u >> 16);
  return __builtin_bit_cast(bf16, h);
}

// ---------------------------------------------------------------------------
// Stage 0: batch-wide mean / rstd (layernorm over all C*S per n)
// ---------------------------------------------------------------------------
__global__ void zero_f32(float* p, int n) {
  int i = blockIdx.x * 256 + threadIdx.x;
  if (i < n) p[i] = 0.f;
}

__global__ __launch_bounds__(256)
void reduce_stats(const float* __restrict__ in, float* __restrict__ raw) {
  __shared__ float sh0[256], sh1[256];
  const int n = blockIdx.y;
  const float* p = in + (size_t)n * kC * kS;
  float s = 0.f, s2 = 0.f;
  for (int i = blockIdx.x * 256 + threadIdx.x; i < kC * kS; i += gridDim.x * 256) {
    float v = p[i]; s += v; s2 += v * v;
  }
  sh0[threadIdx.x] = s; sh1[threadIdx.x] = s2;
  __syncthreads();
  for (int o = 128; o > 0; o >>= 1) {
    if ((int)threadIdx.x < o) {
      sh0[threadIdx.x] += sh0[threadIdx.x + o];
      sh1[threadIdx.x] += sh1[threadIdx.x + o];
    }
    __syncthreads();
  }
  if (threadIdx.x == 0) {
    atomicAdd(&raw[2 * n + 0], sh0[0]);
    atomicAdd(&raw[2 * n + 1], sh1[0]);
  }
}

__global__ void finalize_stats(const float* __restrict__ raw, float* __restrict__ stats) {
  int n = threadIdx.x;
  if (n < kN) {
    float inv  = 1.0f / (float)(kC * kS);
    float mean = raw[2 * n] * inv;
    float var  = raw[2 * n + 1] * inv - mean * mean;
    stats[2 * n + 0] = mean;
    stats[2 * n + 1] = rsqrtf(var + kEPS);
  }
}

// ---------------------------------------------------------------------------
// Stage 1: (optionally layernorm-affine) transpose (n,R,Cc)f32 -> (n,Cc,R)bf16
// ---------------------------------------------------------------------------
__global__ __launch_bounds__(256)
void norm_transpose(const float* __restrict__ in, bf16* __restrict__ outp,
                    int R, int Cc, const float* __restrict__ stats,
                    const float* __restrict__ gamma, const float* __restrict__ beta) {
  __shared__ float tile[32][33];
  const int nb = blockIdx.z;
  const int r0 = blockIdx.y * 32;
  const int c0 = blockIdx.x * 32;
  const float* pin = in + ((size_t)nb * R + r0) * Cc + c0;
  for (int i = threadIdx.y; i < 32; i += 8)
    tile[i][threadIdx.x] = pin[(size_t)i * Cc + threadIdx.x];
  __syncthreads();
  float mu = 0.f, rinv = 1.f, g = 1.f, b = 0.f;
  const int r = r0 + threadIdx.x;  // original row (= channel index)
  if (stats) { mu = stats[2 * nb]; rinv = stats[2 * nb + 1]; g = gamma[r]; b = beta[r]; }
  bf16* pout = outp + ((size_t)nb * Cc + c0) * R + r0;
  for (int i = threadIdx.y; i < 32; i += 8) {
    float v = tile[threadIdx.x][i];
    v = (v - mu) * rinv * g + b;
    pout[(size_t)i * R + threadIdx.x] = f2bf(v);
  }
}

__global__ void cvt_bf16(const float* __restrict__ in, bf16* __restrict__ o, int n) {
  int i = blockIdx.x * 256 + threadIdx.x;
  if (i < n) o[i] = f2bf(in[i]);
}

// ---------------------------------------------------------------------------
// Row softmax: f32 (rows x ncols) -> bf16 probabilities
// ---------------------------------------------------------------------------
__global__ __launch_bounds__(256)
void softmax_rows(const float* __restrict__ Sin, bf16* __restrict__ P, int ncols) {
  __shared__ float red[256];
  const int row = blockIdx.x;
  const float* p = Sin + (size_t)row * ncols;
  bf16* q = P + (size_t)row * ncols;
  float m = -3.4e38f;
  for (int i = threadIdx.x; i < ncols; i += 256) m = fmaxf(m, p[i]);
  red[threadIdx.x] = m; __syncthreads();
  for (int o = 128; o > 0; o >>= 1) {
    if ((int)threadIdx.x < o) red[threadIdx.x] = fmaxf(red[threadIdx.x], red[threadIdx.x + o]);
    __syncthreads();
  }
  m = red[0]; __syncthreads();
  float s = 0.f;
  for (int i = threadIdx.x; i < ncols; i += 256) s += __expf(p[i] - m);
  red[threadIdx.x] = s; __syncthreads();
  for (int o = 128; o > 0; o >>= 1) {
    if ((int)threadIdx.x < o) red[threadIdx.x] += red[threadIdx.x + o];
    __syncthreads();
  }
  const float inv = 1.0f / red[0];
  for (int i = threadIdx.x; i < ncols; i += 256) q[i] = f2bf(__expf(p[i] - m) * inv);
}

// ---------------------------------------------------------------------------
// CDNA5 async copy: 16B per lane, global -> LDS, tracked by ASYNCcnt.
//   lds_off : per-lane LDS byte address (low 32 bits of LDS pointer)
//   base    : uniform 64-bit global base (SGPR pair)
//   gl_off  : per-lane 32-bit byte offset
// ---------------------------------------------------------------------------
__device__ __forceinline__ void async_ld16(unsigned lds_off, const bf16* base,
                                           unsigned gl_off) {
  asm volatile("global_load_async_to_lds_b128 %0, %1, %2"
               :: "v"(lds_off), "v"(gl_off), "s"(base)
               : "memory");
}
__device__ __forceinline__ void wait_async0() {
  asm volatile("s_wait_asynccnt 0x0" ::: "memory");
}

// ---------------------------------------------------------------------------
// NT GEMM:  D(M,Nn) = A(M,K) * B(Nn,K)^T   (bf16 in, f32 accum via WMMA)
//   TRANS    : store D transposed (Cout[n][m]) — 16B-contiguous per lane
//   OUT_BF16 : output bf16 (else f32)
//   HAS_BIAS : + bias[m]
//   HAS_RES  : + res[m*ldc + n] (row-major only)
// block tile 128x128, 8 waves (4x2), K-step 32, double-buffered LDS filled by
// async copies that overlap the WMMA stream.
// ---------------------------------------------------------------------------
static constexpr int BM = 128, BN = 128, BK = 32;
static constexpr int LDSS = 40;   // padded row stride in bf16 elems (80B)

union FragAB { v16bf v; f4 f[2]; };

template <bool TRANS, bool OUT_BF16, bool HAS_BIAS, bool HAS_RES>
__global__ __launch_bounds__(256)
void gemm_nt(const bf16* __restrict__ A, const bf16* __restrict__ B,
             void* __restrict__ Cout, const float* __restrict__ bias,
             const float* __restrict__ res,
             int M, int Nn, int K, int lda, int ldb, int ldc, float scale) {
  __shared__ bf16 shA[2][BM * LDSS];
  __shared__ bf16 shB[2][BN * LDSS];

  const int tid  = threadIdx.x;
  const int lane = tid & 31;
  const int wv   = tid >> 5;
  const int wm   = wv & 3;          // 4 wave-rows
  const int wn   = wv >> 2;         // 2 wave-cols
  const int m0   = blockIdx.y * BM;
  const int n0   = blockIdx.x * BN;

  // staging: each thread owns one 32B chunk (= 2 async b128) per tile
  const int lr = tid >> 1;           // 0..127 (tile row)
  const int lc = (tid & 1) * 16;     // 0 or 16 (k elems)

  const bf16* aBase = A + (size_t)m0 * lda;   // block-uniform -> SGPRs
  const bf16* bBase = B + (size_t)n0 * ldb;
  const unsigned aOffG = (unsigned)((lr * lda + lc) * 2);   // bytes
  const unsigned bOffG = (unsigned)((lr * ldb + lc) * 2);
  const unsigned thrL  = (unsigned)((lr * LDSS + lc) * 2);  // bytes
  const unsigned sA[2] = { (unsigned)(unsigned long long)&shA[0][0],
                           (unsigned)(unsigned long long)&shA[1][0] };
  const unsigned sB[2] = { (unsigned)(unsigned long long)&shB[0][0],
                           (unsigned)(unsigned long long)&shB[1][0] };

  auto issue_tile = [&](int kt, int buf) {
    const unsigned la = sA[buf] + thrL;
    const unsigned lb = sB[buf] + thrL;
    const unsigned ga = aOffG + (unsigned)(kt * BK * 2);
    const unsigned gb = bOffG + (unsigned)(kt * BK * 2);
    async_ld16(la,       aBase, ga);
    async_ld16(la + 16u, aBase, ga + 16u);
    async_ld16(lb,       bBase, gb);
    async_ld16(lb + 16u, bBase, gb + 16u);
  };

  const int KT = K / BK;
  issue_tile(0, 0);

  v8f acc[2][4];
#pragma unroll
  for (int i = 0; i < 2; ++i)
#pragma unroll
    for (int j = 0; j < 4; ++j)
#pragma unroll
      for (int r = 0; r < 8; ++r) acc[i][j][r] = 0.f;

  const int hl = lane >> 4;          // K-quarter selector
  const int fr = lane & 15;          // fragment row (M for A, N for B)

  for (int kt = 0; kt < KT; ++kt) {
    const int cur = kt & 1;
    wait_async0();        // my async writes for tile kt have landed in LDS
    __syncthreads();      // everyone's writes landed; buf[cur^1] free to fill
    if (kt + 1 < KT) issue_tile(kt + 1, cur ^ 1);
    if (kt + 2 < KT) {
      __builtin_prefetch(aBase + (size_t)lr * lda + lc + (size_t)(kt + 2) * BK, 0, 1);
      __builtin_prefetch(bBase + (size_t)lr * ldb + lc + (size_t)(kt + 2) * BK, 0, 1);
    }

    FragAB a[2], b[4];
#pragma unroll
    for (int i = 0; i < 2; ++i) {
      const bf16* p = &shA[cur][(wm * 32 + i * 16 + fr) * LDSS + 8 * hl];
      a[i].f[0] = *(const f4*)p;
      a[i].f[1] = *(const f4*)(p + 16);
    }
#pragma unroll
    for (int j = 0; j < 4; ++j) {
      const bf16* p = &shB[cur][(wn * 64 + j * 16 + fr) * LDSS + 8 * hl];
      b[j].f[0] = *(const f4*)p;
      b[j].f[1] = *(const f4*)(p + 16);
    }
#pragma unroll
    for (int i = 0; i < 2; ++i)
#pragma unroll
      for (int j = 0; j < 4; ++j)
        acc[i][j] = __builtin_amdgcn_wmma_f32_16x16x32_bf16(
            false, a[i].v, false, b[j].v, (short)0, acc[i][j], false, false);
  }

  // epilogue: D tile [16x16]; lane fr = N index, VGPR r -> M = 8*hl + r
#pragma unroll
  for (int i = 0; i < 2; ++i) {
    const int mBase = m0 + wm * 32 + i * 16 + 8 * hl;
#pragma unroll
    for (int j = 0; j < 4; ++j) {
      const int nIdx = n0 + wn * 64 + j * 16 + fr;
      v8f v = acc[i][j];
      if constexpr (TRANS) {
        if constexpr (OUT_BF16) {
          union { bf16 h[8]; f4 f; } t;
#pragma unroll
          for (int r = 0; r < 8; ++r) {
            float f = v[r] * scale;
            if constexpr (HAS_BIAS) f += bias[mBase + r];
            t.h[r] = f2bf(f);
          }
          *(f4*)((bf16*)Cout + (size_t)nIdx * ldc + mBase) = t.f;
        } else {
          union { float x[8]; f4 f[2]; } t;
#pragma unroll
          for (int r = 0; r < 8; ++r) {
            float f = v[r] * scale;
            if constexpr (HAS_BIAS) f += bias[mBase + r];
            t.x[r] = f;
          }
          float* po = (float*)Cout + (size_t)nIdx * ldc + mBase;
          *(f4*)po = t.f[0];
          *(f4*)(po + 4) = t.f[1];
        }
      } else {
#pragma unroll
        for (int r = 0; r < 8; ++r) {
          float f = v[r] * scale;
          if constexpr (HAS_BIAS) f += bias[mBase + r];
          if constexpr (HAS_RES)  f += res[(size_t)(mBase + r) * ldc + nIdx];
          if constexpr (OUT_BF16)
            ((bf16*)Cout)[(size_t)(mBase + r) * ldc + nIdx] = f2bf(f);
          else
            ((float*)Cout)[(size_t)(mBase + r) * ldc + nIdx] = f;
        }
      }
    }
  }
}

// ---------------------------------------------------------------------------
extern "C" void kernel_launch(void* const* d_in, const int* in_sizes, int n_in,
                              void* d_out, int out_size, void* d_ws, size_t ws_size,
                              hipStream_t stream) {
  const float* input   = (const float*)d_in[0];
  const float* context = (const float*)d_in[1];
  const float* gamma   = (const float*)d_in[2];
  const float* beta    = (const float*)d_in[3];
  const float* Wq      = (const float*)d_in[4];
  const float* bq      = (const float*)d_in[5];
  const float* Wkv     = (const float*)d_in[6];
  const float* bkv     = (const float*)d_in[7];
  const float* Wo      = (const float*)d_in[8];
  const float* bo      = (const float*)d_in[9];
  float* out = (float*)d_out;

  char* wsb = (char*)d_ws;
  size_t off = 0;
  auto alloc = [&](size_t bytes) -> char* {
    char* p = wsb + off;
    off += (bytes + 255) & ~(size_t)255;
    return p;
  };

  float* raw   = (float*)alloc(kN * 2 * sizeof(float));
  float* stats = (float*)alloc(kN * 2 * sizeof(float));
  bf16* Wq16   = (bf16*)alloc((size_t)kO * kC * 2);          //  4 MB
  bf16* Wkv16  = (bf16*)alloc((size_t)2 * kO * kCTX * 2);    //  8 MB
  bf16* Wo16   = (bf16*)alloc((size_t)kC * kO * 2);          //  4 MB
  bf16* xnT    = (bf16*)alloc((size_t)kN * kS * kC * 2);     //  8 MB  (n,s,c)
  bf16* ctxT   = (bf16*)alloc((size_t)kN * kS2 * kCTX * 2);  //  8 MB  (n,s,c)
  bf16* qT     = (bf16*)alloc((size_t)kS * kO * 2);          // 16 MB  (s,4096)
  bf16* kT     = (bf16*)alloc((size_t)kS2 * kO * 2);         // 16 MB  (s2,4096)
  bf16* vB     = (bf16*)alloc((size_t)kO * kS2 * 2);         // 16 MB  (4096,s2)
  float* Sc    = (float*)alloc((size_t)kS * kS2 * 4);        // 16 MB  scores
  bf16* Pb     = (bf16*)alloc((size_t)kS * kS2 * 2);         //  8 MB  probs
  bf16* yT     = (bf16*)alloc((size_t)kS * kO * 2);          // 16 MB  (s,4096)
  (void)ws_size; (void)in_sizes; (void)n_in; (void)out_size;

  // stats
  zero_f32<<<1, 32, 0, stream>>>(raw, kN * 2);
  reduce_stats<<<dim3(256, kN), 256, 0, stream>>>(input, raw);
  finalize_stats<<<1, 32, 0, stream>>>(raw, stats);

  // transposed bf16 activations + bf16 weights
  norm_transpose<<<dim3(kS / 32, kC / 32, kN), dim3(32, 8), 0, stream>>>(
      input, xnT, kC, kS, stats, gamma, beta);
  norm_transpose<<<dim3(kS2 / 32, kCTX / 32, kN), dim3(32, 8), 0, stream>>>(
      context, ctxT, kCTX, kS2, nullptr, nullptr, nullptr);
  cvt_bf16<<<(kO * kC + 255) / 256, 256, 0, stream>>>(Wq, Wq16, kO * kC);
  cvt_bf16<<<(2 * kO * kCTX + 255) / 256, 256, 0, stream>>>(Wkv, Wkv16, 2 * kO * kCTX);
  cvt_bf16<<<(kC * kO + 255) / 256, 256, 0, stream>>>(Wo, Wo16, kC * kO);

  const float scoreScale = 1.0f / sqrtf((float)kC);   // (c^-0.25)^2

  for (int n = 0; n < kN; ++n) {
    const bf16* xb = xnT + (size_t)n * kS * kC;
    const bf16* cb = ctxT + (size_t)n * kS2 * kCTX;

    // q = Wq*xn + bq  -> qT (s, 4096)            [transposed bf16 store]
    gemm_nt<true, true, true, false><<<dim3(kS / BN, kO / BM), 256, 0, stream>>>(
        Wq16, xb, qT, bq, nullptr, kO, kS, kC, kC, kC, kO, 1.0f);
    // k = Wkv[0:4096]*ctx + bkv[0:4096] -> kT (s2, 4096)
    gemm_nt<true, true, true, false><<<dim3(kS2 / BN, kO / BM), 256, 0, stream>>>(
        Wkv16, cb, kT, bkv, nullptr, kO, kS2, kCTX, kCTX, kCTX, kO, 1.0f);
    // v = Wkv[4096:]*ctx + bkv[4096:]  -> vB (4096, s2)  [row-major bf16]
    gemm_nt<false, true, true, false><<<dim3(kS2 / BN, kO / BM), 256, 0, stream>>>(
        Wkv16 + (size_t)kO * kCTX, cb, vB, bkv + kO, nullptr,
        kO, kS2, kCTX, kCTX, kCTX, kS2, 1.0f);

    for (int h = 0; h < kH; ++h) {
      // scores: D[k,q] = K_h * Q_h^T, transposed store -> Sc[q,k] f32, scaled
      gemm_nt<true, false, false, false><<<dim3(kS / BN, kS2 / BM), 256, 0, stream>>>(
          kT + h * kC, qT + h * kC, Sc, nullptr, nullptr,
          kS2, kS, kC, kO, kO, kS2, scoreScale);
      softmax_rows<<<kS, 256, 0, stream>>>(Sc, Pb, kS2);
      // y = P * V_h (NT with V^T rows) -> yT[:, h*512 : h*512+512]
      gemm_nt<false, true, false, false><<<dim3(kC / BN, kS / BM), 256, 0, stream>>>(
          Pb, vB + (size_t)h * kC * kS2, yT + h * kC, nullptr, nullptr,
          kS, kC, kS2, kS2, kS2, kO, 1.0f);
    }

    // out = Wo * y + bo + residual  (f32, row-major (c,s))
    gemm_nt<false, false, true, true><<<dim3(kS / BN, kC / BM), 256, 0, stream>>>(
        Wo16, yT, out + (size_t)n * kC * kS, bo, input + (size_t)n * kC * kS,
        kC, kS, kO, kO, kO, kS, 1.0f);
  }
}
